// Encoder_45990509806098
// MI455X (gfx1250) — matrix operand order repaired and checked by reading, not compiled, not run on hardware
//
#include <hip/hip_runtime.h>
#include <hip/hip_bf16.h>

typedef _Float16 f16;
typedef __attribute__((ext_vector_type(16))) _Float16 v16h;
typedef __attribute__((ext_vector_type(8)))  float    v8f;

union ABReg { v16h v; unsigned int u[8]; };

// 32 halfs + 8 pad = 80B row stride: 16B-aligned rows for async b128 copies,
// and 20-dword stride keeps 16-lane gather groups bank-conflict-free.
#define LDS_ST 40

// ---------------------------------------------------------------------------
// CDNA5 async global->LDS copy (ASYNCcnt path, ISA 08_async_tensor).
// vdst VGPR holds the LDS byte address (low 32 bits of generic LDS pointer).
// ---------------------------------------------------------------------------
__device__ __forceinline__ void async_b128(const f16* g, f16* l) {
    unsigned loff = (unsigned)(unsigned long long)(void*)l;
    unsigned long long ga = (unsigned long long)(const void*)g;
    asm volatile("global_load_async_to_lds_b128 %0, %1, off"
                 :: "v"(loff), "v"(ga) : "memory");
}
__device__ __forceinline__ void wait_async0() {
    asm volatile("s_wait_asynccnt 0x0" ::: "memory");
}

// ---------------------------------------------------------------------------
// f32 -> f16 cast
// ---------------------------------------------------------------------------
__global__ __launch_bounds__(256) void cast_f32_f16(const float* __restrict__ in,
                                                    f16* __restrict__ out, int n) {
    int i = blockIdx.x * 256 + threadIdx.x;
    if (i < n) out[i] = (f16)in[i];
}

// ---------------------------------------------------------------------------
// WMMA GEMM: C[M,N] = A[M,K] * B + bias, f16 in, f32 accumulate.
// BT=false: B is [K,N] row-major.  BT=true: B is [N,K] row-major (C = A*B^T).
// Block = 256 threads (8 waves) -> 128x64 tile; wave w owns a 32x32 subtile:
// 4 accumulators, 2 A + 2 B operands, 4 WMMAs per 32-deep k-step.
// Double-buffered LDS; A tile (and BT B tile) staged with async global->LDS
// copies, prefetching tile s+1 under the WMMAs of tile s.
// blockIdx.z batches heads via element strides aZ/bZ/cZ.
// ---------------------------------------------------------------------------
template <typename OutT, bool BIAS, bool RELU, bool BT>
__global__ __launch_bounds__(256)
void gemm_wmma(const f16* __restrict__ A, const f16* __restrict__ B,
               OutT* __restrict__ C, const float* __restrict__ bias,
               int lda, int ldb, int ldc, int K,
               long aZ, long bZ, long cZ) {
    __shared__ f16 sA[2][128 * LDS_ST];   // [m][k]
    __shared__ f16 sB[2][64 * LDS_ST];    // [n][k] (always k-contiguous)

    const int tid  = threadIdx.x;
    const int lane = tid & 31;
    const int w    = tid >> 5;
    const int z    = blockIdx.z;
    A += (long)z * aZ;  B += (long)z * bZ;  C += (long)z * cZ;

    const int n0 = blockIdx.x * 64;
    const int m0 = blockIdx.y * 128;
    const int mo = (w >> 1) * 32;   // wave row offset in tile
    const int no = (w & 1) * 32;    // wave col offset in tile
    const int hl = lane >> 4;       // lane half (ISA 7.12.2 layout)
    const int ml = lane & 15;

    v8f acc00 = {}, acc01 = {}, acc10 = {}, acc11 = {};

    // stage tile at k0 into buffer `buf`
    auto stage = [&](int buf, int k0) {
        // A tile 128x32: 2 async b128 copies per thread (no VGPR round-trip)
#pragma unroll
        for (int ch = 0; ch < 2; ++ch) {
            int chunk = tid + ch * 256;       // 0..511
            int row = chunk >> 2;             // 0..127
            int c0  = (chunk & 3) * 8;        // 0,8,16,24
            async_b128(A + (long)(m0 + row) * lda + (k0 + c0),
                       &sA[buf][row * LDS_ST + c0]);
        }
        if (BT) {  // B is [N,K]: direct copy -> async as well
            int nr = tid >> 2;                // 0..63
            int c0 = (tid & 3) * 8;
            async_b128(B + (long)(n0 + nr) * ldb + (k0 + c0),
                       &sB[buf][nr * LDS_ST + c0]);
        } else {   // B is [K,N]: coalesced read along n, transpose into LDS
            int c   = tid >> 3;               // 0..31
            int nn0 = (tid & 7) * 8;          // 0..56
            uint4 d = *(const uint4*)(B + (long)(k0 + c) * ldb + (n0 + nn0));
            const f16* src = (const f16*)&d;
#pragma unroll
            for (int i = 0; i < 8; ++i) sB[buf][(nn0 + i) * LDS_ST + c] = src[i];
        }
    };

    // prologue: stage first tile
    stage(0, 0);
    wait_async0();
    __syncthreads();

    const int nsteps = K >> 5;
    for (int s = 0; s < nsteps; ++s) {
        const int cur = s & 1;
        const f16* cA = sA[cur];
        const f16* cB = sB[cur];

        // ---- gather operands per ISA 7.12.2 wave32 layouts (from cur) ----
        ABReg a0, a1, b0, b1;
#pragma unroll
        for (int j = 0; j < 8; ++j) {
            // A 16-bit 16x32: VGPR j<4 -> K=2j(+8*half); j>=4 -> K=16+2(j-4)(+8*half)
            int kb = (j < 4 ? 2 * j : 16 + 2 * (j - 4)) + hl * 8;
            a0.u[j] = *(const unsigned int*)(cA + (mo + ml) * LDS_ST + kb);
            a1.u[j] = *(const unsigned int*)(cA + (mo + 16 + ml) * LDS_ST + kb);
            // B 16-bit 32x16: same K order as A (j<4 -> K=2j(+8*half); j>=4 -> K=16+2(j-4)(+8*half))
            int cb = kb;
            b0.u[j] = *(const unsigned int*)(cB + (no + ml) * LDS_ST + cb);
            b1.u[j] = *(const unsigned int*)(cB + (no + 16 + ml) * LDS_ST + cb);
        }

        // ---- prefetch next tile into the other buffer (under the WMMAs) ----
        if (s + 1 < nsteps) stage(cur ^ 1, (s + 1) * 32);

        acc00 = __builtin_amdgcn_wmma_f32_16x16x32_f16(false, a0.v, false, b0.v,
                                                       (short)0, acc00, false, false);
        acc01 = __builtin_amdgcn_wmma_f32_16x16x32_f16(false, a0.v, false, b1.v,
                                                       (short)0, acc01, false, false);
        acc10 = __builtin_amdgcn_wmma_f32_16x16x32_f16(false, a1.v, false, b0.v,
                                                       (short)0, acc10, false, false);
        acc11 = __builtin_amdgcn_wmma_f32_16x16x32_f16(false, a1.v, false, b1.v,
                                                       (short)0, acc11, false, false);

        if (s + 1 < nsteps) wait_async0();   // next tile landed in LDS
        __syncthreads();
    }

    // ---- epilogue: D layout lane = N col, VGPR r -> M = r + 8*half ----
#pragma unroll
    for (int r = 0; r < 8; ++r) {
        int mA = m0 + mo + r + 8 * hl;        // a0 rows
        int mB = mA + 16;                     // a1 rows
        int nA = n0 + no + ml;                // b0 cols
        int nB = nA + 16;                     // b1 cols
        float v00 = acc00[r], v01 = acc01[r], v10 = acc10[r], v11 = acc11[r];
        if (BIAS) {
            float fA = bias[nA], fB = bias[nB];
            v00 += fA; v01 += fB; v10 += fA; v11 += fB;
        }
        if (RELU) {
            v00 = v00 > 0.f ? v00 : 0.f;  v01 = v01 > 0.f ? v01 : 0.f;
            v10 = v10 > 0.f ? v10 : 0.f;  v11 = v11 > 0.f ? v11 : 0.f;
        }
        C[(long)mA * ldc + nA] = (OutT)v00;
        C[(long)mA * ldc + nB] = (OutT)v01;
        C[(long)mB * ldc + nA] = (OutT)v10;
        C[(long)mB * ldc + nB] = (OutT)v11;
    }
}

// ---------------------------------------------------------------------------
// Row softmax with scale + mask, IN-PLACE: reads f32 row [S], writes f16 row
// into the same storage (first half of the row's bytes). One block per row.
// grid = (S rows, H heads); scores layout [h][q][k] f32, ld = S.
// ---------------------------------------------------------------------------
__global__ __launch_bounds__(256)
void softmax_rows(float* __restrict__ sc, const int* __restrict__ mask, int S) {
    __shared__ float buf[256];
    const int q = blockIdx.x, h = blockIdx.y, tid = threadIdx.x;
    float* row = sc + ((long)h * S + q) * S;
    f16*  prow = (f16*)row;

    float v[4]; float mx = -3.4e38f;
#pragma unroll
    for (int i = 0; i < 4; ++i) {
        int c = tid + i * 256;
        float s = row[c] * 0.125f;            // 1/sqrt(64)
        if (mask[c] == 0) s = -1e10f;
        v[i] = s; mx = fmaxf(mx, s);
    }
    buf[tid] = mx; __syncthreads();
    for (int t = 128; t; t >>= 1) { if (tid < t) buf[tid] = fmaxf(buf[tid], buf[tid + t]); __syncthreads(); }
    mx = buf[0]; __syncthreads();

    float e[4]; float sum = 0.f;
#pragma unroll
    for (int i = 0; i < 4; ++i) { e[i] = expf(v[i] - mx); sum += e[i]; }
    buf[tid] = sum; __syncthreads();
    for (int t = 128; t; t >>= 1) { if (tid < t) buf[tid] += buf[tid + t]; __syncthreads(); }
    float inv = 1.0f / buf[0];
    __syncthreads();   // all reads of `row` done before f16 overwrite
#pragma unroll
    for (int i = 0; i < 4; ++i) prow[tid + i * 256] = (f16)(e[i] * inv);
}

// ---------------------------------------------------------------------------
// LayerNorm over rows of 768, one block (256 threads, 3 elems/thread) per row.
// ---------------------------------------------------------------------------
template <typename OutT>
__global__ __launch_bounds__(256)
void layernorm_rows(const float* __restrict__ in, OutT* __restrict__ out,
                    const float* __restrict__ g, const float* __restrict__ b, int Dm) {
    __shared__ float buf[256];
    const long row = blockIdx.x;
    const int tid = threadIdx.x;
    const float* x = in + row * Dm;

    float v[3]; float s = 0.f;
#pragma unroll
    for (int i = 0; i < 3; ++i) { v[i] = x[tid + i * 256]; s += v[i]; }
    buf[tid] = s; __syncthreads();
    for (int t = 128; t; t >>= 1) { if (tid < t) buf[tid] += buf[tid + t]; __syncthreads(); }
    float mu = buf[0] / Dm; __syncthreads();

    float s2 = 0.f;
#pragma unroll
    for (int i = 0; i < 3; ++i) { float d = v[i] - mu; s2 += d * d; }
    buf[tid] = s2; __syncthreads();
    for (int t = 128; t; t >>= 1) { if (tid < t) buf[tid] += buf[tid + t]; __syncthreads(); }
    float rstd = rsqrtf(buf[0] / Dm + 1e-5f);
#pragma unroll
    for (int i = 0; i < 3; ++i) {
        int c = tid + i * 256;
        out[row * Dm + c] = (OutT)((v[i] - mu) * rstd * g[c] + b[c]);
    }
}

// ---------------------------------------------------------------------------
extern "C" void kernel_launch(void* const* d_in, const int* in_sizes, int n_in,
                              void* d_out, int out_size, void* d_ws, size_t ws_size,
                              hipStream_t stream) {
    constexpr int Dm = 768, H = 12, DH = 64, FF = 3072, Bb = 8, S = 1024;
    constexpr long BS = (long)Bb * S;          // 8192 rows

    const float* x    = (const float*)d_in[0];
    const int*   mask = (const int*)  d_in[1];
    const float* Wq = (const float*)d_in[2];  const float* bq = (const float*)d_in[3];
    const float* Wk = (const float*)d_in[4];  const float* bk = (const float*)d_in[5];
    const float* Wv = (const float*)d_in[6];  const float* bv = (const float*)d_in[7];
    const float* Wo = (const float*)d_in[8];  const float* bo = (const float*)d_in[9];
    const float* lg = (const float*)d_in[10]; const float* lb = (const float*)d_in[11];
    const float* W1 = (const float*)d_in[12]; const float* b1 = (const float*)d_in[13];
    const float* W2 = (const float*)d_in[14]; const float* b2 = (const float*)d_in[15];

    char* ws = (char*)d_ws;  size_t off = 0;
    auto alloc = [&](size_t bytes) -> void* {
        void* p = ws + off; off += (bytes + 255) & ~(size_t)255; return p;
    };

    f16* Xh  = (f16*)alloc(BS * Dm * 2);          // aliased later as attn buffer
    f16* Wqh = (f16*)alloc((size_t)Dm * Dm * 2);
    f16* Wkh = (f16*)alloc((size_t)Dm * Dm * 2);
    f16* Wvh = (f16*)alloc((size_t)Dm * Dm * 2);
    f16* Woh = (f16*)alloc((size_t)Dm * Dm * 2);
    f16* W1h = (f16*)alloc((size_t)Dm * FF * 2);
    f16* W2h = (f16*)alloc((size_t)FF * Dm * 2);
    f16* Qh  = (f16*)alloc(BS * Dm * 2);
    f16* Kh  = (f16*)alloc(BS * Dm * 2);
    f16* Vh  = (f16*)alloc(BS * Dm * 2);
    f16* Hh  = (f16*)alloc(BS * Dm * 2);
    // big buffer: per-batch scores [H,S,S] f32 (=50.3MB) reused later as FF1 f16
    void* big = alloc((size_t)H * S * S * 4);
    float* Sc  = (float*)big;
    f16*  FF1h = (f16*)big;
    // f32 buffer: attn_out, later reused for ff2
    float* F32b = (float*)alloc(BS * Dm * 4);

    f16* Attnh = Xh;   // alias: x no longer needed once QKV projections done

    auto cast = [&](const float* src, f16* dst, long n) {
        cast_f32_f16<<<dim3((unsigned)((n + 255) / 256)), dim3(256), 0, stream>>>(src, dst, (int)n);
    };
    cast(x,  Xh,  BS * Dm);
    cast(Wq, Wqh, (long)Dm * Dm);  cast(Wk, Wkh, (long)Dm * Dm);
    cast(Wv, Wvh, (long)Dm * Dm);  cast(Wo, Woh, (long)Dm * Dm);
    cast(W1, W1h, (long)Dm * FF);  cast(W2, W2h, (long)FF * Dm);

    // ---- QKV projections: [8192,768] = Xh * W + b, f16 out ----
    dim3 gProj(Dm / 64, BS / 128, 1);
    gemm_wmma<f16, true, false, false><<<gProj, 256, 0, stream>>>(
        Xh, Wqh, Qh, bq, Dm, Dm, Dm, Dm, 0, 0, 0);
    gemm_wmma<f16, true, false, false><<<gProj, 256, 0, stream>>>(
        Xh, Wkh, Kh, bk, Dm, Dm, Dm, Dm, 0, 0, 0);
    gemm_wmma<f16, true, false, false><<<gProj, 256, 0, stream>>>(
        Xh, Wvh, Vh, bv, Dm, Dm, Dm, Dm, 0, 0, 0);

    // ---- attention, per batch (12 heads batched on grid.z) ----
    for (int b = 0; b < Bb; ++b) {
        const long rowOff = (long)b * S * Dm;
        // scores[h][q][k] = Q_bh (SxDH) * K_bh^T   (BT variant), f32 out
        gemm_wmma<float, false, false, true><<<dim3(S / 64, S / 128, H), 256, 0, stream>>>(
            Qh + rowOff, Kh + rowOff, Sc, nullptr,
            Dm, Dm, S, DH, /*aZ=*/DH, /*bZ=*/DH, /*cZ=*/(long)S * S);
        // softmax rows in-place (f32 -> f16, stride 2*S halfs)
        softmax_rows<<<dim3(S, H), 256, 0, stream>>>(Sc, mask + (long)b * S, S);
        // attn = P (S x S, f16, lda=2S) * V_bh (S x DH) -> Attnh cols h*64..h*64+63
        gemm_wmma<f16, false, false, false><<<dim3(1, S / 128, H), 256, 0, stream>>>(
            (const f16*)Sc, Vh + rowOff, Attnh + rowOff, nullptr,
            2 * S, Dm, Dm, S, /*aZ=*/(long)S * 2 * S, /*bZ=*/DH, /*cZ=*/DH);
    }

    // ---- output projection: attn_out = Attn * Wo + bo, f32 out ----
    gemm_wmma<float, true, false, false><<<gProj, 256, 0, stream>>>(
        Attnh, Woh, F32b, bo, Dm, Dm, Dm, Dm, 0, 0, 0);

    // ---- LN -> f16 ----
    layernorm_rows<f16><<<dim3((unsigned)BS), 256, 0, stream>>>(F32b, Hh, lg, lb, Dm);

    // ---- FF1 = relu(Hh * W1 + b1), f16 out [8192,3072] ----
    gemm_wmma<f16, true, true, false><<<dim3(FF / 64, BS / 128, 1), 256, 0, stream>>>(
        Hh, W1h, FF1h, b1, Dm, FF, FF, Dm, 0, 0, 0);

    // ---- FF2 = FF1 * W2 + b2, f32 out [8192,768] ----
    gemm_wmma<float, true, false, false><<<dim3(Dm / 64, BS / 128, 1), 256, 0, stream>>>(
        FF1h, W2h, F32b, b2, FF, Dm, Dm, FF, 0, 0, 0);

    // ---- final LN -> d_out (f32) ----
    layernorm_rows<float><<<dim3((unsigned)BS), 256, 0, stream>>>(
        F32b, (float*)d_out, lg, lb, Dm);
}
